// gnn_33638183862499
// MI455X (gfx1250) — matrix operand order repaired and checked by reading, not compiled, not run on hardware
//
#include <hip/hip_runtime.h>
#include <hip/hip_bf16.h>

typedef __attribute__((ext_vector_type(16))) _Float16 v16h;
typedef __attribute__((ext_vector_type(8)))  float    v8f;

// ---------------------------------------------------------------- init ------
__global__ void k_init(float* deg, float* red, int n) {
    long long t = blockIdx.x * (long long)blockDim.x + threadIdx.x;
    if (t < n)  deg[t] = 1.0f;      // degree includes self-loop (+1)
    if (t < 64) red[t] = 0.0f;      // reduction scratch (key slot 0 == -inf key)
}

// deg[dst] += 1 per edge
__global__ void k_deg(const int* __restrict__ dst, float* deg, long long E) {
    long long t = blockIdx.x * (long long)blockDim.x + threadIdx.x;
    if (t < E) atomicAdd(&deg[dst[t]], 1.0f);
}

__global__ void k_rsqrt(float* dinv, int n) {
    long long t = blockIdx.x * (long long)blockDim.x + threadIdx.x;
    if (t < n) dinv[t] = rsqrtf(dinv[t]);
}

// ------------------------------------------- layer 1: h1 = pos @ W1 --------
// also fused: a1 = h1 * dinv^2 + b1   (self-loop term + bias)
__global__ void k_l1(const float* __restrict__ pos, const float* __restrict__ W1,
                     const float* __restrict__ b1, const float* __restrict__ dinv,
                     float* __restrict__ h1, float* __restrict__ a1, int n) {
    long long t = blockIdx.x * (long long)blockDim.x + threadIdx.x;
    if (t >= (long long)n * 32) return;
    int i = (int)(t >> 5), f = (int)(t & 31);
    float h = pos[2 * i] * W1[f] + pos[2 * i + 1] * W1[32 + f];
    h1[t] = h;
    float di = dinv[i];
    a1[t] = h * di * di + b1[f];
}

// ------------------------------- edge message scatter (F features/edge) ----
// F=32/16: one wave covers 1-2 edges with lane==feature -> coalesced gather of
// h[src] (128B/64B) and coalesced f32 atomics into agg[dst]. Whole feature
// working set (~40MB) sits in the 192MB L2, so these RMWs run at L2 rate.
template <int F>
__global__ void k_edge(const int* __restrict__ src, const int* __restrict__ dst,
                       const float* __restrict__ dinv, const float* __restrict__ h,
                       float* __restrict__ agg, long long E) {
    long long t = blockIdx.x * (long long)blockDim.x + threadIdx.x;
    if (t >= E * F) return;
    long long e = t / F;
    int f = (int)(t - e * F);
    int s = src[e], d = dst[e];
    float coef = dinv[s] * dinv[d];
    atomicAdd(&agg[(long long)d * F + f], h[(long long)s * F + f] * coef);
}

__global__ void k_relu(float* a, long long n) {
    long long t = blockIdx.x * (long long)blockDim.x + threadIdx.x;
    if (t < n) a[t] = fmaxf(a[t], 0.0f);
}

// ------------------- layer 2 GEMM via WMMA: h2 = x1(Nx32) @ W2(32x16) ------
// One wave32 processes TPW row-tiles: W2 is converted to the f16 B layout
// once, then TPW back-to-back v_wmma_f32_16x16x32_f16 with independent
// accumulators (no D->A/B RAW hazard between them). Full tiles take an
// unguarded straight-line store path.
#define TPW 4
__global__ void k_gemm2_wmma(const float* __restrict__ x1, const float* __restrict__ W2,
                             float* __restrict__ h2, int ngroups, int n) {
    int grp  = (int)((blockIdx.x * (long long)blockDim.x + threadIdx.x) >> 5);
    int lane = threadIdx.x & 31;
    if (grp >= ngroups) return;                 // wave-uniform: EXEC stays all-1s
    int row = lane & 15;                        // lane's M (A) / N (B,C) index
    int kh  = lane >> 4;                        // K-half select

    // B 32x16 f16 layout: lane holds column N=lane&15; halves j: K = j + 16*kh.
    v16h b;
#pragma unroll
    for (int j = 0; j < 16; ++j)
        b[j] = (_Float16)W2[(j + 16 * kh) * 16 + row];

#pragma unroll
    for (int tt = 0; tt < TPW; ++tt) {
        int tile = grp * TPW + tt;
        int base = tile * 16;
        if (base >= n) break;                   // wave-uniform

        // A 16x32 f16 layout (matches B): lanes 0-15 hold K={0..15},
        // lanes 16-31 hold K={16..31} of row M=lane&15.
        int m = base + row;
        const float* arow = x1 + (long long)(m < n ? m : n - 1) * 32;
        v16h a;
#pragma unroll
        for (int j = 0; j < 16; ++j)
            a[j] = (_Float16)arow[j + 16 * kh];

        v8f c = {};
        c = __builtin_amdgcn_wmma_f32_16x16x32_f16(
            /*neg_a=*/false, a, /*neg_b=*/false, b,
            /*c_mod=*/(short)0, c, /*reuse_a=*/false, /*reuse_b=*/false);

        float* out = h2 + (long long)tile * 256;    // 16x16 tile
        if (base + 16 <= n) {                       // full tile: no guards
#pragma unroll
            for (int v = 0; v < 8; ++v)
                out[(v + 8 * kh) * 16 + row] = c[v];
        } else {                                    // ragged tail tile
#pragma unroll
            for (int v = 0; v < 8; ++v) {
                int mm = v + 8 * kh;
                if (base + mm < n) out[mm * 16 + row] = c[v];
            }
        }
    }
}

// a2 = h2 * dinv^2 + b2
__global__ void k_l2init(const float* __restrict__ h2, const float* __restrict__ b2,
                         const float* __restrict__ dinv, float* __restrict__ a2, int n) {
    long long t = blockIdx.x * (long long)blockDim.x + threadIdx.x;
    if (t >= (long long)n * 16) return;
    int i = (int)(t >> 4), f = (int)(t & 15);
    float di = dinv[i];
    a2[t] = h2[t] * di * di + b2[f];
}

// layer 3: h3[i] = dot(x2[i], W3); c[i] = h3*dinv^2 + b3
__global__ void k_l3(const float* __restrict__ x2, const float* __restrict__ W3,
                     const float* __restrict__ b3, const float* __restrict__ dinv,
                     float* __restrict__ h3, float* __restrict__ c, int n) {
    long long t = blockIdx.x * (long long)blockDim.x + threadIdx.x;
    if (t >= n) return;
    const float* xr = x2 + t * 16;
    float s = 0.0f;
#pragma unroll
    for (int k = 0; k < 16; ++k) s += xr[k] * W3[k];
    h3[t] = s;
    float di = dinv[t];
    c[t] = s * di * di + b3[0];
}

// ---------------------------------------------------- softmax over nodes ---
__device__ __forceinline__ unsigned f2key(float v) {
    unsigned b = __float_as_uint(v);
    return b ^ ((b & 0x80000000u) ? 0xFFFFFFFFu : 0x80000000u);
}
__device__ __forceinline__ float key2f(unsigned k) {
    return __uint_as_float(k ^ ((k & 0x80000000u) ? 0x80000000u : 0xFFFFFFFFu));
}

__global__ void k_max(const float* __restrict__ c, unsigned* red, int n) {
    __shared__ unsigned s[256];
    long long t = blockIdx.x * (long long)blockDim.x + threadIdx.x;
    s[threadIdx.x] = (t < n) ? f2key(c[t]) : 0u;
    __syncthreads();
    for (int off = 128; off > 0; off >>= 1) {
        if (threadIdx.x < off) s[threadIdx.x] = max(s[threadIdx.x], s[threadIdx.x + off]);
        __syncthreads();
    }
    if (threadIdx.x == 0) atomicMax(&red[0], s[0]);
}

__global__ void k_sumexp(const float* __restrict__ c, float* red, int n) {
    __shared__ float s[256];
    float mx = key2f(((const unsigned*)red)[0]);
    long long t = blockIdx.x * (long long)blockDim.x + threadIdx.x;
    s[threadIdx.x] = (t < n) ? __expf(c[t] - mx) : 0.0f;
    __syncthreads();
    for (int off = 128; off > 0; off >>= 1) {
        if (threadIdx.x < off) s[threadIdx.x] += s[threadIdx.x + off];
        __syncthreads();
    }
    if (threadIdx.x == 0) atomicAdd(&red[1], s[0]);
}

__global__ void k_choice(const float* __restrict__ c, const float* __restrict__ red,
                         float* __restrict__ out, int n) {
    long long t = blockIdx.x * (long long)blockDim.x + threadIdx.x;
    if (t >= n) return;
    float mx = key2f(((const unsigned*)red)[0]);
    out[t] = __expf(c[t] - mx) / red[1];
}

// ------------------------------------------------- global mean pool (16) ---
__global__ void k_mean(const float* __restrict__ x2, float* red, int n) {
    __shared__ float s[256 * 16];
    float acc[16];
#pragma unroll
    for (int f = 0; f < 16; ++f) acc[f] = 0.0f;
    for (long long i = blockIdx.x * (long long)blockDim.x + threadIdx.x; i < n;
         i += gridDim.x * (long long)blockDim.x) {
        const float* xr = x2 + i * 16;
#pragma unroll
        for (int f = 0; f < 16; ++f) acc[f] += xr[f];
    }
#pragma unroll
    for (int f = 0; f < 16; ++f) s[threadIdx.x * 16 + f] = acc[f];
    __syncthreads();
    for (int off = 128; off > 0; off >>= 1) {
        if (threadIdx.x < off)
#pragma unroll
            for (int f = 0; f < 16; ++f)
                s[threadIdx.x * 16 + f] += s[(threadIdx.x + off) * 16 + f];
        __syncthreads();
    }
    if (threadIdx.x < 16) atomicAdd(&red[8 + threadIdx.x], s[threadIdx.x]);
}

__global__ void k_value(const float* __restrict__ red, const float* __restrict__ Wfc,
                        const float* __restrict__ bfc, float* __restrict__ out, int n) {
    if (blockIdx.x != 0 || threadIdx.x != 0) return;
    float v = 0.0f;
#pragma unroll
    for (int f = 0; f < 16; ++f) v += (red[8 + f] / (float)n) * Wfc[f];
    out[n] = v + bfc[0];
}

// ---------------------------------------------------------------------------
extern "C" void kernel_launch(void* const* d_in, const int* in_sizes, int n_in,
                              void* d_out, int out_size, void* d_ws, size_t ws_size,
                              hipStream_t stream) {
    const float* pos = (const float*)d_in[0];
    const int*   ei  = (const int*)d_in[1];
    const float* W1  = (const float*)d_in[2];
    const float* b1  = (const float*)d_in[3];
    const float* W2  = (const float*)d_in[4];
    const float* b2  = (const float*)d_in[5];
    const float* W3  = (const float*)d_in[6];
    const float* b3  = (const float*)d_in[7];
    const float* Wfc = (const float*)d_in[8];
    const float* bfc = (const float*)d_in[9];

    const int       n = in_sizes[0] / 2;          // pos is [N,2]
    const long long E = in_sizes[1] / 2;          // edge_index is [2,E]
    const int* src = ei;
    const int* dst = ei + E;

    // workspace carve-up (floats): ~ (99*N + 64) * 4 bytes ≈ 40 MB
    float* ws   = (float*)d_ws;
    float* dinv = ws;                       // N   (degree, then rsqrt in place)
    float* h1   = dinv + n;                 // 32N
    float* a1   = h1 + 32LL * n;            // 32N (-> x1 after relu)
    float* h2   = a1 + 32LL * n;            // 16N
    float* a2   = h2 + 16LL * n;            // 16N (-> x2 after relu)
    float* h3   = a2 + 16LL * n;            // N
    float* c    = h3 + n;                   // N
    float* red  = c + n;                    // 64

    float* out = (float*)d_out;             // [0..N-1]=choice, [N]=value

    const int BS = 256;
    auto nb = [](long long work, int bs) { return (unsigned)((work + bs - 1) / bs); };

    // degrees + dinv
    k_init  <<<nb(n, BS), BS, 0, stream>>>(dinv, red, n);
    k_deg   <<<nb(E, BS), BS, 0, stream>>>(dst, dinv, E);
    k_rsqrt <<<nb(n, BS), BS, 0, stream>>>(dinv, n);

    // layer 1 (F=32)
    k_l1      <<<nb(32LL * n, BS), BS, 0, stream>>>(pos, W1, b1, dinv, h1, a1, n);
    k_edge<32><<<nb(E * 32, BS), BS, 0, stream>>>(src, dst, dinv, h1, a1, E);
    k_relu    <<<nb(32LL * n, BS), BS, 0, stream>>>(a1, 32LL * n);

    // layer 2 (F=16), dense part on WMMA (TPW tiles per wave)
    int ntiles  = (n + 15) / 16;
    int ngroups = (ntiles + TPW - 1) / TPW;
    k_gemm2_wmma<<<nb((long long)ngroups * 32, BS), BS, 0, stream>>>(a1, W2, h2, ngroups, n);
    k_l2init  <<<nb(16LL * n, BS), BS, 0, stream>>>(h2, b2, dinv, a2, n);
    k_edge<16><<<nb(E * 16, BS), BS, 0, stream>>>(src, dst, dinv, h2, a2, E);
    k_relu    <<<nb(16LL * n, BS), BS, 0, stream>>>(a2, 16LL * n);

    // layer 3 (F=1) -> logits c
    k_l3     <<<nb(n, BS), BS, 0, stream>>>(a2, W3, b3, dinv, h3, c, n);
    k_edge<1><<<nb(E, BS), BS, 0, stream>>>(src, dst, dinv, h3, c, E);

    // softmax over nodes -> choice
    k_max    <<<nb(n, BS), BS, 0, stream>>>(c, (unsigned*)red, n);
    k_sumexp <<<nb(n, BS), BS, 0, stream>>>(c, red, n);
    k_choice <<<nb(n, BS), BS, 0, stream>>>(c, red, out, n);

    // mean pool + value head
    k_mean  <<<512, BS, 0, stream>>>(a2, red, n);
    k_value <<<1, 64, 0, stream>>>(red, Wfc, bfc, out, n);
}
